// sLSTMBlock_910533067125
// MI455X (gfx1250) — matrix operand order repaired and checked by reading, not compiled, not run on hardware
//
#include <hip/hip_runtime.h>
#include <hip/hip_bf16.h>
#include <math.h>

typedef __attribute__((ext_vector_type(16))) _Float16 v16h;
typedef __attribute__((ext_vector_type(8)))  _Float16 v8h;
typedef __attribute__((ext_vector_type(4)))  _Float16 v4h;
typedef __attribute__((ext_vector_type(8)))  float    v8f;
typedef __attribute__((ext_vector_type(4)))  int      v4i;

#define BATCH 16384
#define HDIM  1024   // = IN = H
#define KSTEPS (HDIM / 32)

#if defined(__gfx1250__) && \
    __has_builtin(__builtin_amdgcn_global_load_async_to_lds_b128) && \
    __has_builtin(__builtin_amdgcn_s_wait_asynccnt)
#define USE_ASYNC 1
#else
#define USE_ASYNC 0
#endif

#if USE_ASYNC
typedef __attribute__((address_space(1))) v4i gbl_v4i;   // global int4
typedef __attribute__((address_space(3))) v4i lds_v4i;   // LDS int4
#endif

template<bool V> struct bool_c { static constexpr bool value = V; };

__device__ __forceinline__ float sigf(float x) {
    return 1.0f / (1.0f + __expf(-x));
}

// 16B global -> LDS copy (async DMA on CDNA5, sync fallback otherwise)
__device__ __forceinline__ void cp16(const _Float16* g, _Float16* l) {
#if USE_ASYNC
    __builtin_amdgcn_global_load_async_to_lds_b128(
        (gbl_v4i*)(void*)g, (lds_v4i*)(void*)l, 0, 0);
#else
    *(uint4*)l = *(const uint4*)g;
#endif
}

template<int N> __device__ __forceinline__ void wait_async() {
#if USE_ASYNC
    __builtin_amdgcn_s_wait_asynccnt(N);
#endif
}

// ---------------------------------------------------------------------------
// LayerNorm of x over IN=1024, also emit f16 copies of x and x_norm.
// ---------------------------------------------------------------------------
__global__ __launch_bounds__(256) void ln_kernel(
    const float* __restrict__ x, const float* __restrict__ gamma,
    const float* __restrict__ beta,
    _Float16* __restrict__ x16, _Float16* __restrict__ xn16) {
    __shared__ float rs0[256];
    __shared__ float rs1[256];
    const int tid = threadIdx.x;
    const size_t row = blockIdx.x;
    const float4 v = ((const float4*)(x + row * HDIM))[tid];
    float s = v.x + v.y + v.z + v.w;
    float q = v.x * v.x + v.y * v.y + v.z * v.z + v.w * v.w;
    rs0[tid] = s; rs1[tid] = q;
    __syncthreads();
    for (int off = 128; off > 0; off >>= 1) {
        if (tid < off) { rs0[tid] += rs0[tid + off]; rs1[tid] += rs1[tid + off]; }
        __syncthreads();
    }
    const float mu  = rs0[0] * (1.0f / HDIM);
    const float var = rs1[0] * (1.0f / HDIM) - mu * mu;
    const float rsg = rsqrtf(var + 1e-5f);
    const float4 g = ((const float4*)gamma)[tid];
    const float4 b = ((const float4*)beta)[tid];
    v4h xo, no;
    xo[0] = (_Float16)v.x; xo[1] = (_Float16)v.y;
    xo[2] = (_Float16)v.z; xo[3] = (_Float16)v.w;
    no[0] = (_Float16)((v.x - mu) * rsg * g.x + b.x);
    no[1] = (_Float16)((v.y - mu) * rsg * g.y + b.y);
    no[2] = (_Float16)((v.z - mu) * rsg * g.z + b.z);
    no[3] = (_Float16)((v.w - mu) * rsg * g.w + b.w);
    *(v4h*)(x16  + row * HDIM + tid * 4) = xo;
    *(v4h*)(xn16 + row * HDIM + tid * 4) = no;
}

// ---------------------------------------------------------------------------
// f32 -> f16 conversion (4 elems/thread)
// ---------------------------------------------------------------------------
__global__ __launch_bounds__(256) void cvt_kernel(
    const float* __restrict__ src, _Float16* __restrict__ dst, int n) {
    const int i = (blockIdx.x * 256 + threadIdx.x) * 4;
    if (i < n) {
        const float4 v = *(const float4*)(src + i);
        v4h o;
        o[0] = (_Float16)v.x; o[1] = (_Float16)v.y;
        o[2] = (_Float16)v.z; o[3] = (_Float16)v.w;
        *(v4h*)(dst + i) = o;
    }
}

// ---------------------------------------------------------------------------
// Fused 4-gate GEMM + sLSTM cell epilogue.
// Grid: (H/64, B/64). Block: 256 threads = 8 waves.
// Wave (wm,wn): rows wm*16..+15, cols wn*32..+31 (two 16x16 WMMA tiles).
// Double-buffered LDS pipeline driven by async global->LDS DMA.
// ---------------------------------------------------------------------------
#define LDA 40  // padded halves per LDS tile row (80 B; kills bank conflicts)

__device__ __forceinline__ v16h ldfrag(const _Float16* p) {
    v8h lo = *(const v8h*)p;
    v8h hi = *(const v8h*)(p + 8);
    return __builtin_shufflevector(lo, hi, 0, 1, 2, 3, 4, 5, 6, 7,
                                           8, 9, 10, 11, 12, 13, 14, 15);
}

__global__ __launch_bounds__(256) void slstm_gemm(
    const _Float16* __restrict__ x16, const _Float16* __restrict__ xn16,
    const _Float16* __restrict__ h16,
    const _Float16* __restrict__ w16,   // 4 x [1024x1024] (z,i,f,o)
    const _Float16* __restrict__ r16,   // 4 x [1024x1024] (z,i,f,o)
    const float* __restrict__ bz, const float* __restrict__ bi,
    const float* __restrict__ bf, const float* __restrict__ bo,
    const float* __restrict__ brz, const float* __restrict__ bri,
    const float* __restrict__ brf, const float* __restrict__ bro,
    const float* __restrict__ gate_init, const float* __restrict__ c_prev,
    float* __restrict__ out_h, float* __restrict__ out_c) {

    __shared__ _Float16 sA0[2][64 * LDA];      // x (phase0) / h (phase1)
    __shared__ _Float16 sA1[2][64 * LDA];      // x_norm (phase0 only)
    __shared__ _Float16 sB[2][4][64 * LDA];    // weight tiles per gate

    const int tid  = threadIdx.x;
    const int lane = tid & 31;
    const int wave = tid >> 5;
    const int wm   = wave & 3;    // row slab (x16 rows)
    const int wn   = wave >> 2;   // col slab (x32 cols)
    const int fn   = lane & 15;   // fragment row/col within tile
    const int kg   = lane >> 4;   // K-group select

    const int mbase = blockIdx.y * 64;
    const int nbase = blockIdx.x * 64;

    // --- accumulators seeded with (b + br) biases (C operand = free bias) ---
    v8f acc[4][2];
    const float* bptr[4][2] = {{bz, brz}, {bi, bri}, {bf, brf}, {bo, bro}};
#pragma unroll
    for (int s = 0; s < 2; ++s) {
        const int col = nbase + wn * 32 + s * 16 + fn;
#pragma unroll
        for (int g = 0; g < 4; ++g)
            acc[g][s] = (v8f)(bptr[g][0][col] + bptr[g][1][col]);
    }

    // --- staging indices (256 threads move 64 rows x 4 x 16B chunks) ---
    const int sr = tid >> 2;                       // tile row 0..63
    const int sc = tid & 3;                        // source 8-half chunk
    const int sp = ((sc & 1) << 1) | (sc >> 1);    // A dest group: G0 G2 G1 G3
    const int ldsA = sr * LDA + sp * 8;            // permuted (A fragments)
    const int ldsB = sr * LDA + sp * 8;            // permuted (B fragments)
    const int arow  = (wm * 16 + fn) * LDA + kg * 16;
    const int brow0 = (wn * 32 + fn) * LDA + kg * 16;   // +16*LDA for s=1

    // Phase runner: HAS_A1 -> gates i,f read A1 (x_norm); else all gates A0.
    auto run_phase = [&](const _Float16* a0g, const _Float16* a1g,
                         const _Float16* wbg, auto tag) {
        constexpr bool HAS_A1  = decltype(tag)::value;
        constexpr int PER_TILE = HAS_A1 ? 6 : 5;    // async ops per thread/tile
        const _Float16* a0p = a0g + (size_t)(mbase + sr) * HDIM + sc * 8;
        const _Float16* a1p = HAS_A1 ? a1g + (size_t)(mbase + sr) * HDIM + sc * 8
                                     : a0p;
        const _Float16* wbp = wbg + (size_t)(nbase + sr) * HDIM + sc * 8;

        auto issue = [&](int kk, int buf) {
            cp16(a0p + kk, &sA0[buf][ldsA]);
            if constexpr (HAS_A1) cp16(a1p + kk, &sA1[buf][ldsA]);
#pragma unroll
            for (int g = 0; g < 4; ++g)
                cp16(wbp + (size_t)g * (HDIM * HDIM) + kk, &sB[buf][g][ldsB]);
        };

        issue(0, 0);                                  // prologue: tile 0
        for (int it = 0; it < KSTEPS; ++it) {
            __syncthreads();                          // buffer-reuse guard
            const int cur = it & 1;
            if (it + 1 < KSTEPS) {
                issue((it + 1) * 32, cur ^ 1);        // overlap next DMA
                wait_async<PER_TILE>();               // retire tile `it`
            } else {
                wait_async<0>();                      // drain
            }
            __syncthreads();                          // tile `it` visible to all

            const v16h a0 = ldfrag(&sA0[cur][arow]);
            v16h a1 = a0;
            if constexpr (HAS_A1) a1 = ldfrag(&sA1[cur][arow]);
#pragma unroll
            for (int s = 0; s < 2; ++s) {
#pragma unroll
                for (int g = 0; g < 4; ++g) {
                    const v16h b = ldfrag(&sB[cur][g][brow0 + s * 16 * LDA]);
                    const v16h& a = (HAS_A1 && (g == 1 || g == 2)) ? a1 : a0;
                    acc[g][s] = __builtin_amdgcn_wmma_f32_16x16x32_f16(
                        false, a, false, b, (short)0, acc[g][s], false, false);
                }
            }
        }
    };

    run_phase(x16, xn16, w16, bool_c<true>{});   // W-phase: z,o<-x ; i,f<-x_norm
    run_phase(h16, h16,  r16, bool_c<false>{});  // R-phase: all gates <- h_prev

    // --- epilogue: gates + cell update; C/D layout: lane->col, vgpr->row ---
#pragma unroll
    for (int s = 0; s < 2; ++s) {
        const int col = nbase + wn * 32 + s * 16 + fn;
        const float gi = gate_init[col];
#pragma unroll
        for (int r = 0; r < 8; ++r) {
            const int row = mbase + wm * 16 + kg * 8 + r;
            const size_t idx = (size_t)row * HDIM + col;
            const float zv = tanhf(acc[0][s][r]);
            const float iv = sigf(acc[1][s][r]);
            const float fv = sigf(acc[2][s][r]);
            const float ov = sigf(acc[3][s][r]);
            const float fe = sigf(gi * fv) + 0.1f;
            const float ct = fe * c_prev[idx] + iv * zv;
            out_c[idx] = ct;
            out_h[idx] = ov * tanhf(ct);
        }
    }
}

// ---------------------------------------------------------------------------
extern "C" void kernel_launch(void* const* d_in, const int* in_sizes, int n_in,
                              void* d_out, int out_size, void* d_ws, size_t ws_size,
                              hipStream_t stream) {
    const float* x         = (const float*)d_in[0];
    const float* h_prev    = (const float*)d_in[1];
    const float* c_prev    = (const float*)d_in[2];
    const float* ln_gamma  = (const float*)d_in[3];
    const float* ln_beta   = (const float*)d_in[4];
    const float* gate_init = (const float*)d_in[5];
    const float* Wz = (const float*)d_in[6];   const float* bz  = (const float*)d_in[7];
    const float* Wi = (const float*)d_in[8];   const float* bi  = (const float*)d_in[9];
    const float* Wf = (const float*)d_in[10];  const float* bf  = (const float*)d_in[11];
    const float* Wo = (const float*)d_in[12];  const float* bo  = (const float*)d_in[13];
    const float* Rz = (const float*)d_in[14];  const float* brz = (const float*)d_in[15];
    const float* Ri = (const float*)d_in[16];  const float* bri = (const float*)d_in[17];
    const float* Rf = (const float*)d_in[18];  const float* brf = (const float*)d_in[19];
    const float* Ro = (const float*)d_in[20];  const float* bro = (const float*)d_in[21];

    const size_t act_bytes = (size_t)BATCH * HDIM * sizeof(_Float16);   // 32 MB
    const size_t mat_elems = (size_t)HDIM * HDIM;                       // 1M
    char* ws = (char*)d_ws;
    _Float16* x16  = (_Float16*)(ws);
    _Float16* xn16 = (_Float16*)(ws + act_bytes);
    _Float16* h16  = (_Float16*)(ws + 2 * act_bytes);
    _Float16* w16  = (_Float16*)(ws + 3 * act_bytes);
    _Float16* r16  = w16 + 4 * mat_elems;

    // 1) layernorm + f16 conversion of x / x_norm
    ln_kernel<<<BATCH, 256, 0, stream>>>(x, ln_gamma, ln_beta, x16, xn16);

    // 2) f16 conversions: h_prev and the 8 weight matrices
    cvt_kernel<<<(BATCH * HDIM) / 1024, 256, 0, stream>>>(h_prev, h16, BATCH * HDIM);
    const float* Ws[4] = {Wz, Wi, Wf, Wo};
    const float* Rs[4] = {Rz, Ri, Rf, Ro};
    for (int g = 0; g < 4; ++g) {
        cvt_kernel<<<(int)(mat_elems / 1024), 256, 0, stream>>>(Ws[g], w16 + g * mat_elems, (int)mat_elems);
        cvt_kernel<<<(int)(mat_elems / 1024), 256, 0, stream>>>(Rs[g], r16 + g * mat_elems, (int)mat_elems);
    }

    // 3) fused WMMA GEMM + sLSTM cell
    float* out_h = (float*)d_out;
    float* out_c = out_h + (size_t)BATCH * HDIM;
    dim3 grid(HDIM / 64, BATCH / 64);
    slstm_gemm<<<grid, 256, 0, stream>>>(x16, xn16, h16, w16, r16,
                                         bz, bi, bf, bo, brz, bri, brf, bro,
                                         gate_init, c_prev, out_h, out_c);
}
